// WanSelfAttention_17042430230737
// MI455X (gfx1250) — matrix operand order repaired and checked by reading, not compiled, not run on hardware
//
#include <hip/hip_runtime.h>
#include <hip/hip_bf16.h>

// ---------------- CDNA5 WMMA plumbing ----------------
typedef __attribute__((ext_vector_type(16))) _Float16 v16h;
typedef __attribute__((ext_vector_type(8)))  float    v8f;
typedef __attribute__((ext_vector_type(4)))  unsigned int v4u;
typedef __attribute__((ext_vector_type(8)))  int    v8i;
typedef __attribute__((ext_vector_type(4)))  int    v4i;

struct F16frag { union { v16h v; uint4 q[2]; }; };

__device__ __forceinline__ v8f zero8() {
  v8f z = {0.f,0.f,0.f,0.f,0.f,0.f,0.f,0.f};
  return z;
}

__device__ __forceinline__ v8f wmma16(v16h a, v16h b, v8f c) {
  // D = A(16x32 f16) * B(32x16 f16) + C(16x16 f32)
  return __builtin_amdgcn_wmma_f32_16x16x32_f16(
      /*neg_a=*/false, a, /*neg_b=*/false, b,
      /*c_mod=*/(short)0, c, /*reuse_a=*/false, /*reuse_b=*/false);
}

// Async global->LDS copy (ASYNCcnt-tracked), 16B per lane per op.
__device__ __forceinline__ void async_b128(unsigned lds, const void* gsrc) {
  asm volatile("global_load_async_to_lds_b128 %0, %1, off"
               :: "v"(lds), "v"(gsrc) : "memory");
}
__device__ __forceinline__ void async_b128_off16(unsigned lds, const void* gsrc) {
  asm volatile("global_load_async_to_lds_b128 %0, %1, off offset:16"
               :: "v"(lds), "v"(gsrc) : "memory");
}
__device__ __forceinline__ void wait_async0() {
  asm volatile("s_wait_asynccnt 0x0" ::: "memory");
}
__device__ __forceinline__ unsigned lds_addr(const void* p) {
  // generic addr low 32 bits == wave-relative LDS offset (flat aperture mapping)
  return (unsigned)(unsigned long long)p;
}

// ---- Tensor Data Mover: 2D tile of f16 rows -> LDS with row padding ----
// Loads `rows` x `row_halves` (f16) from a tensor with dim0 = row_halves,
// dim0_stride = row_halves, into LDS with 4-DWORD pad after every 64 DWORDs
// (i.e. LDS pitch = row_halves + 8 halves; requires row_halves == 128).
// D# layout per CDNA5 ISA. 6-arg builtin (clang-23 form).
__device__ __forceinline__ void tdm_load_rows_f16(unsigned lds, const void* gaddr,
                                                  unsigned rows, unsigned row_halves,
                                                  unsigned tensor_rows) {
  unsigned long long ga = (unsigned long long)gaddr;
  v4u g0;
  g0.x = 1u;                                              // count=1 (valid user D#)
  g0.y = lds;                                             // lds_addr (bytes)
  g0.z = (unsigned)(ga & 0xFFFFFFFFu);                    // global_addr[31:0]
  g0.w = (unsigned)((ga >> 32) & 0x1FFFFFFu) | (2u << 30); // addr[56:32] | type=2
  v8i g1;
  // dword0: wg_mask=0 | data_size=1(2B)<<16 | pad_enable<<20 | pad_interval=5(64dw)<<22 | pad_amount=3(4dw)<<25
  g1[0] = (int)((1u << 16) | (1u << 20) | (5u << 22) | (3u << 25));
  g1[1] = (int)((row_halves & 0xFFFFu) << 16);                                     // tensor_dim0[15:0]
  g1[2] = (int)(((row_halves >> 16) & 0xFFFFu) | ((tensor_rows & 0xFFFFu) << 16)); // dim0 hi | dim1 lo
  g1[3] = (int)(((tensor_rows >> 16) & 0xFFFFu) | ((row_halves & 0xFFFFu) << 16)); // dim1 hi | tile_dim0
  g1[4] = (int)(rows & 0xFFFFu);                                                   // tile_dim1 | tile_dim2=0
  g1[5] = (int)row_halves;                                                         // tensor_dim0_stride lo32
  g1[6] = 0;                                                                       // stride hi | dim1_stride lo
  g1[7] = 0;
  v4i z4 = {0, 0, 0, 0};
  v8i z8 = {0, 0, 0, 0, 0, 0, 0, 0};
  __builtin_amdgcn_tensor_load_to_lds(g0, g1, z4, z4, z8, 0);
}

#define DIM    1536
#define NHEADS 12
#define HD     128

// ---------------- elementwise fp32 -> f16 ----------------
__global__ void cvt16_kernel(const float* __restrict__ in, _Float16* __restrict__ out, int n) {
  int i = blockIdx.x * blockDim.x + threadIdx.x;
  if (i < n) out[i] = (_Float16)in[i];
}

// ---------------- RoPE cos/sin table: [L][64] ----------------
__global__ void rope_tab_kernel(const float* __restrict__ freqs,
                                const long long* __restrict__ gsz,
                                float* __restrict__ cosb, float* __restrict__ sinb,
                                int Ltok) {
  int idx = blockIdx.x * blockDim.x + threadIdx.x;
  if (idx >= Ltok * 64) return;
  int s = idx >> 6, j = idx & 63;
  int H = (int)gsz[1], W = (int)gsz[2];
  int hw = H * W;
  int f = s / hw, rem = s - f * hw;
  int h = rem / W, w = rem - h * W;
  int pos = (j < 22) ? f : ((j < 43) ? h : w);
  float ang = freqs[pos * 64 + j];
  cosb[idx] = cosf(ang);
  sinb[idx] = sinf(ang);
}

// ---------------- generic f16 GEMM: C(f32)[M,N] = A[M,K] * B[K,N] + bias ----------------
// Double-buffered, software-pipelined; A tile via async-to-LDS, B tile transposed in VGPRs.
#define BM 128
#define BN 128
#define BK 32
#define AP 40   // LDS pitch in halves (80B rows, 16B-aligned)
#define BP 40

__global__ __launch_bounds__(256)
void gemm_f16f32_kernel(const _Float16* __restrict__ A, const _Float16* __restrict__ B,
                        const float* __restrict__ bias, float* __restrict__ C,
                        int M, int N, int K) {
  __shared__ _Float16 As[2][BM * AP];   // [m][k]
  __shared__ _Float16 Bt[2][BN * BP];   // [n][k] (transposed tile)

  const int tid  = threadIdx.x;
  const int wave = tid >> 5;
  const int lane = tid & 31;
  const int m0 = blockIdx.y * BM;
  const int n0 = blockIdx.x * BN;
  const int wm = wave & 3;    // row group: 32 rows each
  const int wn = wave >> 2;   // col group: 64 cols each

  const int halfsel = lane >> 4;
  const int lrow    = lane & 15;
  const int kbaseA  = halfsel * 8;
  const int kbaseB  = halfsel * 16;

  const int rA = tid >> 1, segA = tid & 1;   // A: 128 rows x 2 segs of 16 halves
  const int kkB = tid >> 3, segB = tid & 7;  // B: 32 k-rows x 8 segs of 16 cols
  const _Float16* gA = A + (size_t)(m0 + rA) * K + segA * 16;
  const _Float16* gB = B + (size_t)kkB * N + n0 + segB * 16;

  v8f acc[2][4];
#pragma unroll
  for (int i = 0; i < 2; ++i)
#pragma unroll
    for (int j = 0; j < 4; ++j) acc[i][j] = zero8();

  uint4 b0, b1;  // pipelined B registers

  // ---- prologue: tile 0 ----
  {
    const uint4* s = (const uint4*)(gB);
    b0 = s[0]; b1 = s[1];
    unsigned lds = lds_addr(&As[0][rA * AP + segA * 16]);
    async_b128(lds, gA);
    async_b128_off16(lds, gA);
    _Float16 tmp[16];
    *(uint4*)(tmp) = b0; *(uint4*)(tmp + 8) = b1;
#pragma unroll
    for (int i = 0; i < 16; ++i) Bt[0][(segB * 16 + i) * BP + kkB] = tmp[i];
  }
  wait_async0();
  __syncthreads();

  const int nkt = K / BK;
  int buf = 0;
  for (int t = 0; t < nkt; ++t) {
    const int nxt = t + 1;
    if (nxt < nkt) {  // issue next tile early: B to regs, A async into other buffer
      const uint4* s = (const uint4*)(gB + (size_t)nxt * BK * N);
      b0 = s[0]; b1 = s[1];
      unsigned lds = lds_addr(&As[buf ^ 1][rA * AP + segA * 16]);
      const _Float16* src = gA + nxt * BK;
      async_b128(lds, src);
      async_b128_off16(lds, src);
    }

    // compute current tile
    F16frag af[2], bf[4];
#pragma unroll
    for (int i = 0; i < 2; ++i) {
      const _Float16* p = &As[buf][(wm * 32 + i * 16 + lrow) * AP + kbaseA];
      af[i].q[0] = *(const uint4*)(p);
      af[i].q[1] = *(const uint4*)(p + 16);
    }
#pragma unroll
    for (int j = 0; j < 4; ++j) {
      const _Float16* p = &Bt[buf][(wn * 64 + j * 16 + lrow) * BP + kbaseA];
      bf[j].q[0] = *(const uint4*)(p);
      bf[j].q[1] = *(const uint4*)(p + 16);
    }
#pragma unroll
    for (int i = 0; i < 2; ++i)
#pragma unroll
      for (int j = 0; j < 4; ++j)
        acc[i][j] = wmma16(af[i].v, bf[j].v, acc[i][j]);

    if (nxt < nkt) {  // transpose-store next B tile
      _Float16 tmp[16];
      *(uint4*)(tmp) = b0; *(uint4*)(tmp + 8) = b1;
#pragma unroll
      for (int i = 0; i < 16; ++i) Bt[buf ^ 1][(segB * 16 + i) * BP + kkB] = tmp[i];
    }
    wait_async0();
    __syncthreads();
    buf ^= 1;
  }

  // epilogue
#pragma unroll
  for (int i = 0; i < 2; ++i) {
    int mbase = m0 + wm * 32 + i * 16 + halfsel * 8;
#pragma unroll
    for (int j = 0; j < 4; ++j) {
      int n = n0 + wn * 64 + j * 16 + lrow;
      float bv = bias[n];
#pragma unroll
      for (int r = 0; r < 8; ++r)
        C[(size_t)(mbase + r) * N + n] = acc[i][j][r] + bv;
    }
  }
}

// ---------------- RMSNorm + RoPE + pack f16 into [head][L][128] ----------------
__global__ __launch_bounds__(256)
void norm_rope_kernel(const float* __restrict__ Y, const float* __restrict__ g,
                      const float* __restrict__ cosb, const float* __restrict__ sinb,
                      _Float16* __restrict__ out, int Ltok) {
  int s = blockIdx.x;
  const float* y = Y + (size_t)s * DIM;
  __shared__ float red[256];
  float ss = 0.f;
  for (int i = threadIdx.x; i < DIM; i += 256) { float v = y[i]; ss += v * v; }
  red[threadIdx.x] = ss;
  __syncthreads();
  for (int w = 128; w > 0; w >>= 1) {
    if (threadIdx.x < w) red[threadIdx.x] += red[threadIdx.x + w];
    __syncthreads();
  }
  float rs = rsqrtf(red[0] * (1.0f / DIM) + 1e-6f);
  for (int p = threadIdx.x; p < (NHEADS * HD / 2); p += 256) {
    int n = p >> 6, j = p & 63;
    float a  = y[2 * p]     * rs * g[2 * p];
    float b  = y[2 * p + 1] * rs * g[2 * p + 1];
    float c  = cosb[(size_t)s * 64 + j];
    float si = sinb[(size_t)s * 64 + j];
    float o0 = a * c - b * si;
    float o1 = a * si + b * c;
    _Float16* dst = out + ((size_t)n * Ltok + s) * HD + 2 * j;
    dst[0] = (_Float16)o0;
    dst[1] = (_Float16)o1;
  }
}

// ---------------- V pack f16 into [head][L][128] ----------------
__global__ void v_pack_kernel(const float* __restrict__ Y, _Float16* __restrict__ vh, int Ltok) {
  int idx = blockIdx.x * blockDim.x + threadIdx.x;
  if (idx >= Ltok * DIM) return;
  int s = idx / DIM, e = idx - s * DIM;
  int n = e >> 7, d = e & 127;
  vh[((size_t)n * Ltok + s) * HD + d] = (_Float16)Y[idx];
}

// ---------------- flash attention: per (head, 128-query block) ----------------
#define AQ 128
#define AK 64
#define QP 136   // Q/K LDS pitch (halves) == TDM pad: 64 DWORDs + 4 DWORD pad
#define VP 72    // P / V^T LDS pitch
#define SP 68    // S f32 pitch

__global__ __launch_bounds__(256)
void attn_kernel(const _Float16* __restrict__ Q, const _Float16* __restrict__ Kp,
                 const _Float16* __restrict__ Vp, const long long* __restrict__ seq_lens,
                 _Float16* __restrict__ Out, int Ltok) {
  __shared__ _Float16 Qs[AQ * QP];
  __shared__ _Float16 Ks[AK * QP];
  __shared__ _Float16 Vt[HD * VP];
  __shared__ float    Ss[AQ * SP];
  __shared__ _Float16 Ps[AQ * VP];
  __shared__ float    mstat[AQ], lstat[AQ], alpha[AQ];

  const int tid  = threadIdx.x;
  const int wave = tid >> 5;
  const int lane = tid & 31;
  const int head = blockIdx.y;
  const int q0   = blockIdx.x * AQ;
  const int seqlen = (int)seq_lens[0];

  const int halfsel = lane >> 4;
  const int lrow    = lane & 15;
  const int kbaseA  = halfsel * 8;
  const int kbaseB  = halfsel * 16;

  const _Float16* Qh = Q  + (size_t)head * Ltok * HD;
  const _Float16* Kh = Kp + (size_t)head * Ltok * HD;
  const _Float16* Vh = Vp + (size_t)head * Ltok * HD;

  // TDM: DMA Q tile 128x128 f16 -> Qs with native row padding (wave 0 issues)
  if (wave == 0) {
    tdm_load_rows_f16(lds_addr(Qs), Qh + (size_t)q0 * HD, AQ, HD, (unsigned)Ltok);
  }
  if (tid < AQ) { mstat[tid] = -3.0e38f; lstat[tid] = 0.f; }

  v8f oacc[8];
#pragma unroll
  for (int t = 0; t < 8; ++t) oacc[t] = zero8();
  if (wave == 0) __builtin_amdgcn_s_wait_tensorcnt(0);
  __syncthreads();

  const int nkt = Ltok / AK;
  for (int kt = 0; kt < nkt; ++kt) {
    const int kb = kt * AK;
    __syncthreads();  // prior-iter Ks/Vt/Ps consumers done
    // TDM: DMA K tile 64x128 f16 -> Ks (wave 0 issues)
    if (wave == 0) {
      tdm_load_rows_f16(lds_addr(Ks), Kh + (size_t)kb * HD, AK, HD, (unsigned)(Ltok - kb));
    }
    { // V tile 64x128 -> transposed Vt[d][key]
      int r = tid >> 2, seg = tid & 3;
      const _Float16* src = Vh + (size_t)(kb + r) * HD + seg * 32;
      _Float16 tmp[32];
#pragma unroll
      for (int i = 0; i < 4; ++i) *(uint4*)(tmp + 8 * i) = ((const uint4*)src)[i];
#pragma unroll
      for (int i = 0; i < 32; ++i) Vt[(seg * 32 + i) * VP + r] = tmp[i];
    }
    if (wave == 0) __builtin_amdgcn_s_wait_tensorcnt(0);
    __syncthreads();

    // S = Q * K^T (this wave: 16 rows x 64 cols), K-dim 128 -> 4 WMMA steps
    v8f sacc[4];
#pragma unroll
    for (int j = 0; j < 4; ++j) sacc[j] = zero8();
#pragma unroll
    for (int ks = 0; ks < 4; ++ks) {
      F16frag af;
      const _Float16* pa = Qs + (wave * 16 + lrow) * QP + ks * 32 + kbaseA;
      af.q[0] = *(const uint4*)(pa);
      af.q[1] = *(const uint4*)(pa + 16);
#pragma unroll
      for (int j = 0; j < 4; ++j) {
        F16frag bf;
        const _Float16* pb = Ks + (j * 16 + lrow) * QP + ks * 32 + kbaseA;
        bf.q[0] = *(const uint4*)(pb);
        bf.q[1] = *(const uint4*)(pb + 16);
        sacc[j] = wmma16(af.v, bf.v, sacc[j]);
      }
    }
    const float scl = 0.08838834764831845f;  // 1/sqrt(128)
#pragma unroll
    for (int j = 0; j < 4; ++j)
#pragma unroll
      for (int r = 0; r < 8; ++r)
        Ss[(wave * 16 + halfsel * 8 + r) * SP + j * 16 + lrow] = sacc[j][r] * scl;
    __syncthreads();

    // online softmax: one thread per query row
    if (tid < AQ) {
      int row = tid;
      int valid = seqlen - kb; if (valid > AK) valid = AK; if (valid < 0) valid = 0;
      float mo = mstat[row], mx = mo;
      for (int j = 0; j < AK; ++j) {
        float v = (j < valid) ? Ss[row * SP + j] : -3.0e38f;
        mx = fmaxf(mx, v);
      }
      float al = __expf(mo - mx);
      float sum = 0.f;
      for (int j = 0; j < AK; ++j) {
        float p = (j < valid) ? __expf(Ss[row * SP + j] - mx) : 0.f;
        Ps[row * VP + j] = (_Float16)p;
        sum += p;
      }
      mstat[row] = mx;
      lstat[row] = lstat[row] * al + sum;
      alpha[row] = al;
    }
    __syncthreads();

    // rescale running O, then O += P * V (K-dim 64 -> 2 WMMA steps)
    float a8[8];
#pragma unroll
    for (int r = 0; r < 8; ++r) a8[r] = alpha[wave * 16 + halfsel * 8 + r];
#pragma unroll
    for (int t = 0; t < 8; ++t)
#pragma unroll
      for (int r = 0; r < 8; ++r) oacc[t][r] *= a8[r];
#pragma unroll
    for (int ks = 0; ks < 2; ++ks) {
      F16frag af;
      const _Float16* pa = Ps + (wave * 16 + lrow) * VP + ks * 32 + kbaseA;
      af.q[0] = *(const uint4*)(pa);
      af.q[1] = *(const uint4*)(pa + 16);
#pragma unroll
      for (int t = 0; t < 8; ++t) {
        F16frag bf;
        const _Float16* pb = Vt + (t * 16 + lrow) * VP + ks * 32 + kbaseA;
        bf.q[0] = *(const uint4*)(pb);
        bf.q[1] = *(const uint4*)(pb + 16);
        oacc[t] = wmma16(af.v, bf.v, oacc[t]);
      }
    }
  }

  __syncthreads();
  if (tid < AQ) lstat[tid] = 1.f / fmaxf(lstat[tid], 1e-30f);
  __syncthreads();
  float i8[8];
#pragma unroll
  for (int r = 0; r < 8; ++r) i8[r] = lstat[wave * 16 + halfsel * 8 + r];
#pragma unroll
  for (int t = 0; t < 8; ++t)
#pragma unroll
    for (int r = 0; r < 8; ++r) {
      int row = q0 + wave * 16 + halfsel * 8 + r;
      int col = head * HD + t * 16 + lrow;
      Out[(size_t)row * DIM + col] = (_Float16)(oacc[t][r] * i8[r]);
    }
}

// ---------------- host launcher ----------------
extern "C" void kernel_launch(void* const* d_in, const int* in_sizes, int n_in,
                              void* d_out, int out_size, void* d_ws, size_t ws_size,
                              hipStream_t stream) {
  const float*     x     = (const float*)d_in[0];
  const long long* seq   = (const long long*)d_in[1];
  const long long* gsz   = (const long long*)d_in[2];
  const float*     freqs = (const float*)d_in[3];
  const float*     Wq    = (const float*)d_in[4];
  const float*     bq    = (const float*)d_in[5];
  const float*     Wk    = (const float*)d_in[6];
  const float*     bk    = (const float*)d_in[7];
  const float*     Wv    = (const float*)d_in[8];
  const float*     bv    = (const float*)d_in[9];
  const float*     Wo    = (const float*)d_in[10];
  const float*     bo    = (const float*)d_in[11];
  const float*     gq    = (const float*)d_in[12];
  const float*     gk    = (const float*)d_in[13];

  const int L = in_sizes[0] / DIM;

  char* ws = (char*)d_ws;
  size_t off = 0;
  auto carve = [&](size_t bytes) -> void* {
    void* p = ws + off;
    off = (off + bytes + 255) & ~(size_t)255;
    return p;
  };
  _Float16* xh    = (_Float16*)carve((size_t)L * DIM * 2);
  _Float16* wqh   = (_Float16*)carve((size_t)DIM * DIM * 2);
  _Float16* wkh   = (_Float16*)carve((size_t)DIM * DIM * 2);
  _Float16* wvh   = (_Float16*)carve((size_t)DIM * DIM * 2);
  _Float16* woh   = (_Float16*)carve((size_t)DIM * DIM * 2);
  float*    y32   = (float*)carve((size_t)L * DIM * 4);
  _Float16* qh    = (_Float16*)carve((size_t)L * DIM * 2);
  _Float16* kh    = (_Float16*)carve((size_t)L * DIM * 2);
  _Float16* vh    = (_Float16*)carve((size_t)L * DIM * 2);
  _Float16* attnh = (_Float16*)carve((size_t)L * DIM * 2);
  float*    cosb  = (float*)carve((size_t)L * 64 * 4);
  float*    sinb  = (float*)carve((size_t)L * 64 * 4);

  const int nx = L * DIM, nw = DIM * DIM;
  cvt16_kernel<<<(nx + 255) / 256, 256, 0, stream>>>(x, xh, nx);
  cvt16_kernel<<<(nw + 255) / 256, 256, 0, stream>>>(Wq, wqh, nw);
  cvt16_kernel<<<(nw + 255) / 256, 256, 0, stream>>>(Wk, wkh, nw);
  cvt16_kernel<<<(nw + 255) / 256, 256, 0, stream>>>(Wv, wvh, nw);
  cvt16_kernel<<<(nw + 255) / 256, 256, 0, stream>>>(Wo, woh, nw);
  rope_tab_kernel<<<(L * 64 + 255) / 256, 256, 0, stream>>>(freqs, gsz, cosb, sinb, L);

  dim3 gg(DIM / BN, L / BM);
  gemm_f16f32_kernel<<<gg, 256, 0, stream>>>(xh, wqh, bq, y32, L, DIM, DIM);
  norm_rope_kernel<<<L, 256, 0, stream>>>(y32, gq, cosb, sinb, qh, L);
  gemm_f16f32_kernel<<<gg, 256, 0, stream>>>(xh, wkh, bk, y32, L, DIM, DIM);
  norm_rope_kernel<<<L, 256, 0, stream>>>(y32, gk, cosb, sinb, kh, L);
  gemm_f16f32_kernel<<<gg, 256, 0, stream>>>(xh, wvh, bv, y32, L, DIM, DIM);
  v_pack_kernel<<<(nx + 255) / 256, 256, 0, stream>>>(y32, vh, L);

  dim3 ga(L / AQ, NHEADS);
  attn_kernel<<<ga, 256, 0, stream>>>(qh, kh, vh, seq, attnh, L);

  gemm_f16f32_kernel<<<gg, 256, 0, stream>>>(attnh, woh, bo, (float*)d_out, L, DIM, DIM);
}
